// GPSTransformer_21869973471849
// MI455X (gfx1250) — matrix operand order repaired and checked by reading, not compiled, not run on hardware
//
#include <hip/hip_runtime.h>
#include <hip/hip_bf16.h>
#include <math.h>

// ---------------- model/data dims (fixed by the reference) ----------------
#define L_    4
#define C_    256
#define PE_   32
#define IN_   64
#define OUT_  64
#define H_    8
#define DH_   32
#define B_    64
#define NPG_  256
#define N_    (B_ * NPG_)        // 16384
#define E_    (N_ * 16)          // 262144
#define SCALE_ 0.17677669529663687f  // 1/sqrt(32)

typedef __attribute__((ext_vector_type(16))) __bf16 v16bf;
typedef __attribute__((ext_vector_type(8)))  __bf16 v8bf;
typedef __attribute__((ext_vector_type(4)))  __bf16 v4bf;
typedef __attribute__((ext_vector_type(8)))  float  v8f;

__device__ __forceinline__ v8f wmma_bf16(v16bf a, v16bf b, v8f c) {
  // (neg_a, A, neg_b, B, c_mod, C, reuse_a, reuse_b)
  return __builtin_amdgcn_wmma_f32_16x16x32_bf16(false, a, false, b, (short)0, c,
                                                 false, false);
}

__device__ __forceinline__ v16bf cat8(v8bf lo, v8bf hi) {
  return __builtin_shufflevector(lo, hi, 0, 1, 2, 3, 4, 5, 6, 7,
                                 8, 9, 10, 11, 12, 13, 14, 15);
}

__device__ __forceinline__ v16bf frag_from_f4_scaled(float4 a, float4 b,
                                                     float4 c, float4 d,
                                                     float s) {
  v16bf f;
  f[0]=(__bf16)(a.x*s);  f[1]=(__bf16)(a.y*s);  f[2]=(__bf16)(a.z*s);  f[3]=(__bf16)(a.w*s);
  f[4]=(__bf16)(b.x*s);  f[5]=(__bf16)(b.y*s);  f[6]=(__bf16)(b.z*s);  f[7]=(__bf16)(b.w*s);
  f[8]=(__bf16)(c.x*s);  f[9]=(__bf16)(c.y*s);  f[10]=(__bf16)(c.z*s); f[11]=(__bf16)(c.w*s);
  f[12]=(__bf16)(d.x*s); f[13]=(__bf16)(d.y*s); f[14]=(__bf16)(d.z*s); f[15]=(__bf16)(d.w*s);
  return f;
}

__device__ __forceinline__ v16bf frag_from_f4(float4 a, float4 b, float4 c, float4 d) {
  v16bf f;
  f[0]=(__bf16)a.x;  f[1]=(__bf16)a.y;  f[2]=(__bf16)a.z;  f[3]=(__bf16)a.w;
  f[4]=(__bf16)b.x;  f[5]=(__bf16)b.y;  f[6]=(__bf16)b.z;  f[7]=(__bf16)b.w;
  f[8]=(__bf16)c.x;  f[9]=(__bf16)c.y;  f[10]=(__bf16)c.z; f[11]=(__bf16)c.w;
  f[12]=(__bf16)d.x; f[13]=(__bf16)d.y; f[14]=(__bf16)d.z; f[15]=(__bf16)d.w;
  return f;
}

// Fragment K-index maps used here (16-bit operands, wave32; not the ISA table order):
//   A (16xK): lanes 0-15 hold K=0..15, lanes 16-31 K=16..31 -> one 32B run.
//   B (Kx16): lanes 0-15 hold K=0..15, lanes 16-31 K=16..31 -> one 32B run; this file's order, not ISA's.

// =====================================================================
// WMMA bf16 GEMM:  Cout[r, c] = ACT( (A[r,:]+A2[r,:]) @ W + bias )
// Block tile 256(M) x 64(N); 8 waves, each wave 32x64 -> 8 WMMA / k-step.
// Software pipelined: next k-tile staged in registers during WMMAs.
// HAS_A2 / ACT are compile-time -> branch-free load loop and epilogue.
// =====================================================================
template <bool HAS_A2, int ACT>
__global__ __launch_bounds__(256) void gemm_bf16_kernel(
    const float* __restrict__ A, const float* __restrict__ A2,
    const float* __restrict__ W, const float* __restrict__ bias,
    float* __restrict__ Cout, int K, int Ncols) {
  __shared__ __align__(16) __bf16 As[256][32];    // 16 KB, row-major
  __shared__ __align__(16) __bf16 BsT[64][40];    // 5 KB, transposed + pad

  const int tid     = threadIdx.x;
  const int lane    = tid & 31;
  const int wv      = tid >> 5;       // 0..7
  const int ln      = lane & 15;
  const int half    = lane >> 4;
  const int rowBase = blockIdx.x * 256;
  const int colBase = blockIdx.y * 64;
  const bool fullB  = (colBase + 64 <= Ncols);

  const v8f zero = {0.f, 0.f, 0.f, 0.f, 0.f, 0.f, 0.f, 0.f};
  v8f acc[2][4];
#pragma unroll
  for (int a = 0; a < 2; ++a)
#pragma unroll
    for (int nt = 0; nt < 4; ++nt) acc[a][nt] = zero;

  float4 aReg[8], bReg[2];

  auto loadStage = [&](int k0) {
#pragma unroll
    for (int it = 0; it < 8; ++it) {
      int i = tid * 4 + it * 1024;
      int r = i >> 5, c = i & 31;
      size_t gi = (size_t)(rowBase + r) * K + k0 + c;
      float4 v = *(const float4*)&A[gi];
      if (HAS_A2) {
        float4 u = *(const float4*)&A2[gi];
        v.x += u.x; v.y += u.y; v.z += u.z; v.w += u.w;
      }
      aReg[it] = v;
    }
#pragma unroll
    for (int it = 0; it < 2; ++it) {
      int i = tid * 4 + it * 1024;
      int r = i >> 6, c = i & 63;
      if (fullB) {
        bReg[it] = *(const float4*)&W[(size_t)(k0 + r) * Ncols + colBase + c];
      } else {
        float t[4];
#pragma unroll
        for (int q = 0; q < 4; ++q) {
          int gc = colBase + c + q;
          t[q] = (gc < Ncols) ? W[(size_t)(k0 + r) * Ncols + gc] : 0.f;
        }
        bReg[it] = make_float4(t[0], t[1], t[2], t[3]);
      }
    }
  };

  auto storeStage = [&]() {
#pragma unroll
    for (int it = 0; it < 8; ++it) {
      int i = tid * 4 + it * 1024;
      int r = i >> 5, c = i & 31;
      v4bf b;
      b[0] = (__bf16)aReg[it].x; b[1] = (__bf16)aReg[it].y;
      b[2] = (__bf16)aReg[it].z; b[3] = (__bf16)aReg[it].w;
      *(v4bf*)&As[r][c] = b;
    }
#pragma unroll
    for (int it = 0; it < 2; ++it) {
      int i = tid * 4 + it * 1024;
      int r = i >> 6, c = i & 63;
      BsT[c + 0][r] = (__bf16)bReg[it].x;
      BsT[c + 1][r] = (__bf16)bReg[it].y;
      BsT[c + 2][r] = (__bf16)bReg[it].z;
      BsT[c + 3][r] = (__bf16)bReg[it].w;
    }
  };

  loadStage(0);
  for (int k0 = 0; k0 < K; k0 += 32) {
    storeStage();
    __syncthreads();
    if (k0 + 32 < K) loadStage(k0 + 32);   // next tile in flight during WMMAs

    const int ar0 = wv * 32 + ln;
    const int ar1 = wv * 32 + 16 + ln;
    v16bf af0 = cat8(*(const v8bf*)&As[ar0][half * 16],
                     *(const v8bf*)&As[ar0][half * 16 + 8]);
    v16bf af1 = cat8(*(const v8bf*)&As[ar1][half * 16],
                     *(const v8bf*)&As[ar1][half * 16 + 8]);
#pragma unroll
    for (int nt = 0; nt < 4; ++nt) {
      int n = nt * 16 + ln;
      v16bf bf = cat8(*(const v8bf*)&BsT[n][half * 16],
                      *(const v8bf*)&BsT[n][half * 16 + 8]);
      acc[0][nt] = wmma_bf16(af0, bf, acc[0][nt]);
      acc[1][nt] = wmma_bf16(af1, bf, acc[1][nt]);
    }
    __syncthreads();
  }

  // epilogue: bias + activation + store (C/D: M = r + 8*half, N = ln)
#pragma unroll
  for (int a = 0; a < 2; ++a)
#pragma unroll
    for (int nt = 0; nt < 4; ++nt) {
      int gc = colBase + nt * 16 + ln;
      if (gc < Ncols) {
        float bv = bias[gc];
#pragma unroll
        for (int r = 0; r < 8; ++r) {
          int gr = rowBase + wv * 32 + a * 16 + r + half * 8;
          float v = acc[a][nt][r] + bv;
          if constexpr (ACT == 1) v = fmaxf(v, 0.f);
          if constexpr (ACT == 2) v = (v >= 0.f) ? v : 0.01f * v;
          Cout[(size_t)gr * Ncols + gc] = v;
        }
      }
    }
}

// =====================================================================
// GINE edge branch:  agg[dst] += relu(h[src] + edge_attr @ W + b)
// 64 threads per edge, 4 channels per thread; scatter via f32 atomics (L2).
// =====================================================================
__global__ __launch_bounds__(256) void gine_edge_kernel(
    const float* __restrict__ h, const float* __restrict__ ea,
    const float* __restrict__ W, const float* __restrict__ bias,
    const int* __restrict__ src, const int* __restrict__ dst,
    float* __restrict__ agg) {
  int tid = blockIdx.x * 256 + threadIdx.x;
  int e = tid >> 6;
  if (e >= E_) return;
  int c = (tid & 63) << 2;

  float4 acc = *(const float4*)(bias + c);
  const float* eap = ea + (size_t)e * 16;
#pragma unroll
  for (int k = 0; k < 16; ++k) {
    float a = eap[k];
    float4 w = *(const float4*)(W + (size_t)k * C_ + c);
    acc.x += a * w.x; acc.y += a * w.y; acc.z += a * w.z; acc.w += a * w.w;
  }
  int s = src[e], d = dst[e];
  float4 hv = *(const float4*)(h + (size_t)s * C_ + c);
  float m0 = fmaxf(acc.x + hv.x, 0.f);
  float m1 = fmaxf(acc.y + hv.y, 0.f);
  float m2 = fmaxf(acc.z + hv.z, 0.f);
  float m3 = fmaxf(acc.w + hv.w, 0.f);
  float* ap = agg + (size_t)d * C_ + c;
  atomicAdd(ap + 0, m0);
  atomicAdd(ap + 1, m1);
  atomicAdd(ap + 2, m2);
  atomicAdd(ap + 3, m3);
}

// =====================================================================
// BatchNorm (training-mode, stats over node axis). M and residual count
// are compile-time: no integer division, branch-free inner loops.
// =====================================================================
template <int M, int NR>
__global__ __launch_bounds__(256) void bn_stats_kernel(
    const float* __restrict__ y, const float* __restrict__ r1,
    const float* __restrict__ r2, float* __restrict__ stats) {
  int c = threadIdx.x;
  if (M < 256 && c >= M) return;
  size_t row0 = (size_t)blockIdx.x * 256;
  float s = 0.f, s2 = 0.f;
  for (int i = 0; i < 256; ++i) {
    size_t idx = (row0 + i) * M + c;
    float v = y[idx];
    if (NR >= 1) v += r1[idx];
    if (NR >= 2) v += r2[idx];
    s += v; s2 += v * v;
  }
  atomicAdd(&stats[c], s);
  atomicAdd(&stats[M + c], s2);
}

template <int M, int NR>
__global__ __launch_bounds__(256) void bn_apply_kernel(
    const float* __restrict__ y, const float* __restrict__ r1,
    const float* __restrict__ r2, const float* __restrict__ stats,
    const float* __restrict__ gam, const float* __restrict__ bet,
    float* __restrict__ z, int ostride, int ocol) {
  size_t idx = (size_t)blockIdx.x * 256 + threadIdx.x;
  if (idx >= (size_t)N_ * M) return;
  int c = (int)(idx % M);          // M is constexpr -> mul/shift
  size_t row = idx / M;
  const float invN = 1.0f / (float)N_;
  float mu  = stats[c] * invN;
  float var = stats[M + c] * invN - mu * mu;
  float v = y[idx];
  if (NR >= 1) v += r1[idx];
  if (NR >= 2) v += r2[idx];
  z[row * ostride + ocol + c] =
      (v - mu) * rsqrtf(var + 1e-5f) * gam[c] + bet[c];
}

// =====================================================================
// Multi-head attention: one wave per (q-tile of 16, head, graph).
// DH=32 == WMMA K. SCALE folded into the Q fragment, so the WMMA score
// result goes straight to LDS (no post-WMMA VALU chain -> fewer hazard
// NOPs). Shuffle softmax, probs bf16, V staged transposed for b128 frags.
// =====================================================================
__global__ __launch_bounds__(32) void attn_kernel(
    const float* __restrict__ qkv, float* __restrict__ omid) {
  __shared__ __align__(16) float  sS[16][NPG_];   // 16 KB scores
  __shared__ __align__(16) __bf16 sP[16][NPG_];   // 8 KB probabilities
  __shared__ __align__(16) __bf16 sVt[DH_][40];   // 2.5 KB V chunk, [dh][key]

  const int qt   = blockIdx.x;   // 0..15
  const int head = blockIdx.y;   // 0..7
  const int g    = blockIdx.z;   // 0..63
  const int lane = threadIdx.x;
  const int ln   = lane & 15;
  const int half = lane >> 4;
  const int base = g * NPG_;

  // Q fragment (A layout), built once; 1/sqrt(DH) pre-folded into Q
  const float* qp = qkv + (size_t)(base + qt * 16 + ln) * 768 + head * DH_;
  v16bf qf = frag_from_f4_scaled(((const float4*)(qp + half * 16))[0],
                                 ((const float4*)(qp + half * 16))[1],
                                 ((const float4*)(qp + half * 16 + 8))[0],
                                 ((const float4*)(qp + half * 16 + 8))[1],
                                 SCALE_);

  // ---- scores: S = (Q*SCALE) K^T over 16 key tiles ----
  for (int kt = 0; kt < 16; ++kt) {
    // B fragment of K^T: col = key (kt*16+ln), K-run dh = 16*half..+15
    const float* kp = qkv + (size_t)(base + kt * 16 + ln) * 768 + 256 +
                      head * DH_ + half * 16;
    v16bf kf = frag_from_f4(((const float4*)kp)[0], ((const float4*)kp)[1],
                            ((const float4*)kp)[2], ((const float4*)kp)[3]);
    v8f s = {0.f, 0.f, 0.f, 0.f, 0.f, 0.f, 0.f, 0.f};
    s = wmma_bf16(qf, kf, s);
#pragma unroll
    for (int r = 0; r < 8; ++r)
      sS[r + half * 8][kt * 16 + ln] = s[r];
  }
  __syncthreads();

  // ---- softmax per q-row: contiguous 8-col slice per lane ----
  for (int r = 0; r < 16; ++r) {
    float4 v0 = *(const float4*)&sS[r][lane * 8];
    float4 v1 = *(const float4*)&sS[r][lane * 8 + 4];
    float e[8] = {v0.x, v0.y, v0.z, v0.w, v1.x, v1.y, v1.z, v1.w};
    float m = -1e30f;
#pragma unroll
    for (int i = 0; i < 8; ++i) m = fmaxf(m, e[i]);
    for (int off = 16; off; off >>= 1) m = fmaxf(m, __shfl_xor(m, off, 32));
    float sum = 0.f;
#pragma unroll
    for (int i = 0; i < 8; ++i) { e[i] = __expf(e[i] - m); sum += e[i]; }
    for (int off = 16; off; off >>= 1) sum += __shfl_xor(sum, off, 32);
    float inv = 1.0f / sum;
    v8bf pv;
#pragma unroll
    for (int i = 0; i < 8; ++i) pv[i] = (__bf16)(e[i] * inv);
    *(v8bf*)&sP[r][lane * 8] = pv;
  }
  __syncthreads();

  // ---- O = P @ V over 8 key chunks of 32 ----
  v8f acc0 = {0.f, 0.f, 0.f, 0.f, 0.f, 0.f, 0.f, 0.f};
  v8f acc1 = acc0;
  for (int ct = 0; ct < 8; ++ct) {
    // stage V chunk transposed: lane owns key = ct*32+lane (row of 32 dh)
    const float* vp = qkv + (size_t)(base + ct * 32 + lane) * 768 + 512 + head * DH_;
#pragma unroll
    for (int i = 0; i < 8; ++i) {
      float4 t = ((const float4*)vp)[i];
      sVt[i * 4 + 0][lane] = (__bf16)t.x;
      sVt[i * 4 + 1][lane] = (__bf16)t.y;
      sVt[i * 4 + 2][lane] = (__bf16)t.z;
      sVt[i * 4 + 3][lane] = (__bf16)t.w;
    }
    __syncthreads();
    // P fragment (A layout): one 32B run in sP row, K = 16*half..+15
    v16bf pf = cat8(*(const v8bf*)&sP[ln][ct * 32 + half * 16],
                    *(const v8bf*)&sP[ln][ct * 32 + half * 16 + 8]);
    // V fragments (B layout): one 32B run per dh column
    v16bf vf0 = cat8(*(const v8bf*)&sVt[ln][half * 16],
                     *(const v8bf*)&sVt[ln][half * 16 + 8]);
    v16bf vf1 = cat8(*(const v8bf*)&sVt[16 + ln][half * 16],
                     *(const v8bf*)&sVt[16 + ln][half * 16 + 8]);
    acc0 = wmma_bf16(pf, vf0, acc0);
    acc1 = wmma_bf16(pf, vf1, acc1);
    __syncthreads();
  }
#pragma unroll
  for (int r = 0; r < 8; ++r) {
    size_t row = (size_t)(base + qt * 16 + r + half * 8);
    omid[row * C_ + head * DH_ + ln]      = acc0[r];
    omid[row * C_ + head * DH_ + 16 + ln] = acc1[r];
  }
}

// =====================================================================
// Host orchestration
// =====================================================================
extern "C" void kernel_launch(void* const* d_in, const int* in_sizes, int n_in,
                              void* d_out, int out_size, void* d_ws, size_t ws_size,
                              hipStream_t stream) {
  (void)in_sizes; (void)n_in; (void)out_size; (void)ws_size;
  const float* x   = (const float*)d_in[0];
  const float* pe  = (const float*)d_in[1];
  const float* ea  = (const float*)d_in[2];
  const float* encW = (const float*)d_in[3];
  const float* encB = (const float*)d_in[4];
  const float* inNg = (const float*)d_in[5];
  const float* inNb = (const float*)d_in[6];
  const float* peNg = (const float*)d_in[7];
  const float* peNb = (const float*)d_in[8];
  const float* gew = (const float*)d_in[9];
  const float* geb = (const float*)d_in[10];
  const float* gmw = (const float*)d_in[11];
  const float* gmb = (const float*)d_in[12];
  const float* aiw = (const float*)d_in[13];
  const float* aib = (const float*)d_in[14];
  const float* aow = (const float*)d_in[15];
  const float* aob = (const float*)d_in[16];
  const float* m1w = (const float*)d_in[17];
  const float* m1b = (const float*)d_in[18];
  const float* m2w = (const float*)d_in[19];
  const float* m2b = (const float*)d_in[20];
  const float* n1g = (const float*)d_in[21];
  const float* n1b = (const float*)d_in[22];
  const float* n2g = (const float*)d_in[23];
  const float* n2b = (const float*)d_in[24];
  const float* n3g = (const float*)d_in[25];
  const float* n3b = (const float*)d_in[26];
  const float* og  = (const float*)d_in[27];
  const float* ob  = (const float*)d_in[28];
  const float* prg = (const float*)d_in[29];
  const float* prb = (const float*)d_in[30];
  const float* d1w = (const float*)d_in[31];
  const float* d1b = (const float*)d_in[32];
  const float* d2w = (const float*)d_in[33];
  const float* d2b = (const float*)d_in[34];
  const int*   ei  = (const int*)d_in[35];
  const int* src = ei;
  const int* dst = ei + E_;

  const size_t NC = (size_t)N_ * C_;
  float* p = (float*)d_ws;
  float* h      = p; p += NC;
  float* agg    = p; p += NC;
  float* locpre = p; p += NC;
  float* loc    = p; p += NC;
  float* qkv    = p; p += (size_t)N_ * 768;
  float* omid   = p; p += NC;
  float* obuf   = p; p += NC;
  float* glob   = p; p += NC;
  float* mid    = p; p += (size_t)N_ * 512;
  float* fbuf   = p; p += NC;
  float* tmp    = p; p += NC;
  float* yenc   = p; p += (size_t)N_ * 224;
  float* dbuf1  = p; p += NC;
  float* stats  = p; p += 512;

  // Only the four (HAS_A2, ACT) combinations actually used are instantiated.
  auto gemm_n0 = [&](const float* A, const float* W, const float* bias,
                     float* Cout, int K, int Ncols) {
    dim3 grid(N_ / 256, (Ncols + 63) / 64);
    gemm_bf16_kernel<false, 0><<<grid, 256, 0, stream>>>(A, nullptr, W, bias, Cout, K, Ncols);
  };
  auto gemm_n2 = [&](const float* A, const float* W, const float* bias,
                     float* Cout, int K, int Ncols) {
    dim3 grid(N_ / 256, (Ncols + 63) / 64);
    gemm_bf16_kernel<false, 2><<<grid, 256, 0, stream>>>(A, nullptr, W, bias, Cout, K, Ncols);
  };
  auto gemm_a1 = [&](const float* A, const float* A2, const float* W,
                     const float* bias, float* Cout, int K, int Ncols) {
    dim3 grid(N_ / 256, (Ncols + 63) / 64);
    gemm_bf16_kernel<true, 1><<<grid, 256, 0, stream>>>(A, A2, W, bias, Cout, K, Ncols);
  };
  auto gemm_a2 = [&](const float* A, const float* A2, const float* W,
                     const float* bias, float* Cout, int K, int Ncols) {
    dim3 grid(N_ / 256, (Ncols + 63) / 64);
    gemm_bf16_kernel<true, 2><<<grid, 256, 0, stream>>>(A, A2, W, bias, Cout, K, Ncols);
  };

  auto bn = [&](const float* y, const float* r1, const float* r2,
                const float* g, const float* b, float* z, int M,
                int ostride, int ocol) {
    hipMemsetAsync(stats, 0, 2 * 256 * sizeof(float), stream);
    unsigned ablocks = (unsigned)(((size_t)N_ * M + 255) / 256);
    if (M == 256) {
      if (r2) {
        bn_stats_kernel<256, 2><<<N_ / 256, 256, 0, stream>>>(y, r1, r2, stats);
        bn_apply_kernel<256, 2><<<ablocks, 256, 0, stream>>>(y, r1, r2, stats, g, b, z, ostride, ocol);
      } else if (r1) {
        bn_stats_kernel<256, 1><<<N_ / 256, 256, 0, stream>>>(y, r1, r2, stats);
        bn_apply_kernel<256, 1><<<ablocks, 256, 0, stream>>>(y, r1, r2, stats, g, b, z, ostride, ocol);
      } else {
        bn_stats_kernel<256, 0><<<N_ / 256, 256, 0, stream>>>(y, r1, r2, stats);
        bn_apply_kernel<256, 0><<<ablocks, 256, 0, stream>>>(y, r1, r2, stats, g, b, z, ostride, ocol);
      }
    } else if (M == 224) {
      bn_stats_kernel<224, 0><<<N_ / 256, 256, 0, stream>>>(y, r1, r2, stats);
      bn_apply_kernel<224, 0><<<ablocks, 256, 0, stream>>>(y, r1, r2, stats, g, b, z, ostride, ocol);
    } else {
      bn_stats_kernel<32, 0><<<N_ / 256, 256, 0, stream>>>(y, r1, r2, stats);
      bn_apply_kernel<32, 0><<<ablocks, 256, 0, stream>>>(y, r1, r2, stats, g, b, z, ostride, ocol);
    }
  };

  // ---- encoder: h = concat(bn(lrelu(x@encW+b)), bn(pe)) ----
  gemm_n2(x, encW, encB, yenc, IN_, 224);
  bn(yenc, nullptr, nullptr, inNg, inNb, h, 224, C_, 0);
  bn(pe,   nullptr, nullptr, peNg, peNb, h, PE_, C_, 224);

  for (int l = 0; l < L_; ++l) {
    // GINE local branch
    hipMemsetAsync(agg, 0, NC * sizeof(float), stream);
    gine_edge_kernel<<<E_ / 4, 256, 0, stream>>>(
        h, ea, gew + (size_t)l * 16 * C_, geb + (size_t)l * C_, src, dst, agg);
    gemm_a2(h, agg, gmw + (size_t)l * C_ * C_, gmb + (size_t)l * C_, locpre,
            C_, C_);
    bn(locpre, h, nullptr, n1g + l * C_, n1b + l * C_, loc, C_, C_, 0);

    // global attention branch
    gemm_n0(h, aiw + (size_t)l * C_ * 768, aib + (size_t)l * 768, qkv, C_, 768);
    attn_kernel<<<dim3(16, H_, B_), 32, 0, stream>>>(qkv, omid);
    gemm_n0(omid, aow + (size_t)l * C_ * C_, aob + (size_t)l * C_, obuf, C_, C_);
    bn(obuf, h, nullptr, n2g + l * C_, n2b + l * C_, glob, C_, C_, 0);

    // combine + FFN + layer norms (out = loc+glob fused into A-load / BN)
    gemm_a1(loc, glob, m1w + (size_t)l * C_ * 512, m1b + (size_t)l * 512, mid,
            C_, 512);
    gemm_n0(mid, m2w + (size_t)l * 512 * C_, m2b + (size_t)l * C_, fbuf,
            512, C_);
    bn(fbuf, loc, glob, n3g + l * C_, n3b + l * C_, tmp, C_, C_, 0);
    bn(tmp, nullptr, nullptr, og + l * C_, ob + l * C_, h, C_, C_, 0);
  }

  // ---- decoder ----
  bn(h, nullptr, nullptr, prg, prb, tmp, C_, C_, 0);
  gemm_n2(tmp, d1w, d1b, dbuf1, C_, C_);
  gemm_n0(dbuf1, d2w, d2b, (float*)d_out, C_, OUT_);
}
